// NetworkBasic_53180285059449
// MI455X (gfx1250) — matrix operand order repaired and checked by reading, not compiled, not run on hardware
//
#include <hip/hip_runtime.h>

typedef __attribute__((ext_vector_type(16))) _Float16 v16h;
typedef __attribute__((ext_vector_type(8)))  float    v8f;

#define B_   8
#define CIN_ 2
#define H_   64
#define W_   64
#define T_   200
#define HW_  (H_*W_)
#define OC_  8
#define H2_  128
#define W2_  128

// decay constants (tau_sr = tau_ref = {1,2,4})
#define D_SR1  0.36787944117144233f  // exp(-1/1)
#define D_SR2  0.60653065971263342f  // exp(-1/2)
#define D_SR3  0.77880078307140487f  // exp(-1/4)
#define D_REF1 0.36787944117144233f
#define D_REF2 0.60653065971263342f
#define D_REF3 0.77880078307140487f
#define THETA1 30.0f
#define THETA2 50.0f
#define THETA3 100.0f

// ---------------------------------------------------------------------------
// K0: pre-pack conv weights into WMMA B-fragment lane layout.
// B (16-bit, 32x16 KxN), wave32: lane = {Khalf = lane>>4, N = lane&15},
// VGPR v packs K = 2v + 16*Khalf (+32*frag), elements [2v]=K, [2v+1]=K+1.
// Stored as [nfrag][32 lanes][16 halves] so each conv lane reads 32B contiguous.
// ---------------------------------------------------------------------------
__global__ void pack_weights(const float* __restrict__ w1,
                             const float* __restrict__ w2,
                             _Float16* __restrict__ b1f,   // [2][32][16]
                             _Float16* __restrict__ b2f) { // [3][32][16]
  int tid = threadIdx.x;
  if (tid < 64) {                     // conv1: K = 2*5*5 = 50 (pad 64)
    int frag = tid >> 5, lane = tid & 31;
    int n = lane & 15, khalf = lane >> 4;
    for (int v = 0; v < 8; ++v) {
      int kb = 32 * frag + 16 * khalf + 2 * v;
      for (int e = 0; e < 2; ++e) {
        int k = kb + e;
        float val = 0.0f;
        if (n < OC_ && k < 50) {
          int ci = k / 25, rem = k % 25, ky = rem / 5, kx = rem % 5;
          val = w1[((n * 2 + ci) * 5 + ky) * 5 + kx];
        }
        b1f[(frag * 32 + lane) * 16 + 2 * v + e] = (_Float16)val;
      }
    }
  } else if (tid < 160) {             // conv2: K = 8*3*3 = 72 (pad 96)
    int id = tid - 64;
    int frag = id >> 5, lane = id & 31;
    int n = lane & 15, khalf = lane >> 4;
    for (int v = 0; v < 8; ++v) {
      int kb = 32 * frag + 16 * khalf + 2 * v;
      for (int e = 0; e < 2; ++e) {
        int k = kb + e;
        float val = 0.0f;
        if (n < OC_ && k < 72) {
          int ci = k / 9, rem = k % 9, ky = rem / 3, kx = rem % 3;
          val = w2[((n * 8 + ci) * 3 + ky) * 3 + kx];
        }
        b2f[(frag * 32 + lane) * 16 + 2 * v + e] = (_Float16)val;
      }
    }
  }
}

// ---------------------------------------------------------------------------
// K1: psp1 scan over T; input [B,C,H,W,T] (T fastest) -> psp1 fp16 [B*T, C, HW]
// ---------------------------------------------------------------------------
__global__ void psp1_scan(const float* __restrict__ in, _Float16* __restrict__ psp1h) {
  int idx = blockIdx.x * blockDim.x + threadIdx.x;    // over B*C*HW
  if (idx >= B_ * CIN_ * HW_) return;
  int p = idx % HW_;
  int c = (idx / HW_) % CIN_;
  int b = idx / (HW_ * CIN_);
  const float* src = in + (size_t)idx * T_;           // contiguous T per pixel
  float a = 0.0f;
  for (int t = 0; t < T_; ++t) {
    if ((t & 31) == 0) __builtin_prefetch(src + t + 64, 0, 1);  // global_prefetch_b8
    a = D_SR1 * a + src[t];
    psp1h[(((size_t)(b * T_ + t)) * CIN_ + c) * HW_ + p] = (_Float16)a;
  }
}

// ---------------------------------------------------------------------------
// K2/K4: implicit-GEMM conv via WMMA with LDS-staged input.
// Block = 256 threads (8 waves) = one (b,t) plane x 2 image rows (128 px).
// Stage (2 + 2*PAD) rows x CIN x 64 fp16 into LDS (coalesced 128B row loads,
// halo rows zero-filled), barrier, then each wave gathers its 16-pixel x
// 16-outchannel tile from LDS and runs NFRAG v_wmma_f32_16x16x32_f16.
// A (16-bit 16x32 MxK): lane = {Ahalf = lane>>4, M = lane&15},
//   VGPR v holds K pairs: K=2v+16*Ahalf (+32*frag), same order as packed B.
// K index decomposes as k = ci*KK*KK + ky*KK + kx (zero-padded past CIN*KK*KK).
// ---------------------------------------------------------------------------
template <int CIN, int KK, int PAD, int NFRAG>
__global__ void conv_wmma(const _Float16* __restrict__ inh,  // [B*T, CIN, HW]
                          const _Float16* __restrict__ bfr,  // [NFRAG][32][16]
                          float* __restrict__ uout) {        // [B*T, OC_, HW]
  constexpr int KSQ   = KK * KK;
  constexpr int KTOT  = CIN * KSQ;
  constexpr int ROWS  = 2 + 2 * PAD;          // staged rows incl. halo
  constexpr int NSTG  = ROWS * CIN * W_;      // staged halves

  __shared__ _Float16 smem[NSTG];

  int bt = blockIdx.x >> 5;                   // plane (b*T + t)
  int h0 = (blockIdx.x & 31) * 2;             // first of 2 rows handled
  const _Float16* plane = inh + (size_t)bt * CIN * HW_;

  // ---- stage input rows (+halo) into LDS, zero-fill out-of-range rows ----
  for (int i = threadIdx.x; i < NSTG; i += 256) {
    int wcol = i & 63;
    int rest = i >> 6;                        // = r*CIN + ci
    int ci = rest % CIN;
    int r  = rest / CIN;
    int hh = h0 - PAD + r;
    _Float16 v = (_Float16)0.0f;
    if (hh >= 0 && hh < H_) v = plane[ci * HW_ + hh * W_ + wcol];
    smem[i] = v;
  }
  __syncthreads();

  int lane   = threadIdx.x & 31;
  int waveid = threadIdx.x >> 5;
  int lrow   = waveid >> 2;                   // 0 or 1: row within block tile
  int wbase  = (waveid & 3) * 16;             // 16-px column tile
  int pbase  = (h0 + lrow) * W_ + wbase;      // global pixel base of tile

  // B fragments: one contiguous 32B load per lane per fragment
  v16h bf[NFRAG];
#pragma unroll
  for (int f = 0; f < NFRAG; ++f) {
    const _Float16* bp = bfr + (f * 32 + lane) * 16;
#pragma unroll
    for (int e = 0; e < 16; ++e) bf[f][e] = bp[e];
  }

  int m = lane & 15;
  int ahalf = lane >> 4;
  int w = wbase + m;                          // pixel column for this lane's M-row

  v8f acc = {};
#pragma unroll
  for (int f = 0; f < NFRAG; ++f) {
    v16h a = {};
#pragma unroll
    for (int v = 0; v < 8; ++v) {
      int kb = 2 * v + 16 * ahalf + 32 * f;
#pragma unroll
      for (int e = 0; e < 2; ++e) {
        int k = kb + e;
        _Float16 val = (_Float16)0.0f;
        if (k < KTOT) {
          int ci  = k / KSQ;
          int rem = k - ci * KSQ;
          int ky  = rem / KK;
          int kx  = rem - ky * KK;
          int ww = w + kx - PAD;
          if (ww >= 0 && ww < W_)             // row always staged; only W check
            val = smem[((lrow + ky) * CIN + ci) * W_ + ww];
        }
        a[2 * v + e] = val;
      }
    }
    acc = __builtin_amdgcn_wmma_f32_16x16x32_f16(false, a, false, bf[f],
                                                 (short)0, acc, false, false);
  }

  // D (f32 16x16): lane = {Dhalf, N}; VGPR r -> M = r + 8*Dhalf
  int n = lane & 15;
  int dhalf = lane >> 4;
  if (n < OC_) {
    float* up = uout + ((size_t)bt * OC_ + n) * HW_;
#pragma unroll
    for (int r = 0; r < 8; ++r) up[pbase + r + 8 * dhalf] = acc[r];
  }
}

// ---------------------------------------------------------------------------
// K3/K5: fused spike scan (hard threshold + leaky refractory) + next psp scan.
// Reads u fp32, writes next-layer psp fp16; spikes themselves never stored.
// ---------------------------------------------------------------------------
__global__ void spike_psp(const float* __restrict__ u, _Float16* __restrict__ pout,
                          float theta, float dref, float dsr) {
  int idx = blockIdx.x * blockDim.x + threadIdx.x;    // over B*OC_*HW
  if (idx >= B_ * OC_ * HW_) return;
  int p = idx % HW_;
  int o = (idx / HW_) % OC_;
  int b = idx / (HW_ * OC_);
  float r = 0.0f, a = 0.0f;
  for (int t = 0; t < T_; ++t) {
    size_t off = (((size_t)(b * T_ + t)) * OC_ + o) * HW_ + p;
    if ((t & 7) == 0)
      __builtin_prefetch(u + off + (size_t)8 * OC_ * HW_, 0, 1);  // global_prefetch_b8
    float v = u[off] + r;
    float s = (v >= theta) ? 1.0f : 0.0f;
    r = dref * r - theta * s;
    a = dsr * a + s;
    pout[off] = (_Float16)a;
  }
}

// ---------------------------------------------------------------------------
// K6: fused convT(2x2,s=2) + bilinear 2x upsample of psp1 + spike(theta=100).
// convT "VALID" padding (pad=k-1, no kernel flip) => kernel index (1-dy,1-dx).
// Bilinear: half-pixel centers, edge clamped.
// Output layout [B,2,H2,W2,T] (T fastest), hard spikes as f32.
// ---------------------------------------------------------------------------
__global__ void out_kernel(const _Float16* __restrict__ psp3h, // [B*T, 8, HW]
                           const _Float16* __restrict__ psp1h, // [B*T, 2, HW]
                           const float* __restrict__ wup,      // [2][8][2][2]
                           float* __restrict__ out) {
  int idx = blockIdx.x * blockDim.x + threadIdx.x;    // over B*2*H2*W2
  if (idx >= B_ * 2 * H2_ * W2_) return;
  int x = idx & 127;
  int y = (idx >> 7) & 127;
  int o = (idx >> 14) & 1;
  int b = idx >> 15;
  int h = y >> 1, w = x >> 1;
  int dy = y & 1, dx = x & 1;

  float wu[8];
#pragma unroll
  for (int i = 0; i < 8; ++i)
    wu[i] = wup[((o * 8 + i) * 2 + (1 - dy)) * 2 + (1 - dx)];

  // bilinear neighbors at src coord y/2 - 0.25
  int iy0 = (dy == 0) ? (h - 1) : h;
  int ix0 = (dx == 0) ? (w - 1) : w;
  float wy0 = (dy == 0) ? 0.25f : 0.75f;
  float wx0 = (dx == 0) ? 0.25f : 0.75f;
  int cy0 = iy0 < 0 ? 0 : iy0;
  int cy1 = (iy0 + 1) > 63 ? 63 : (iy0 + 1);
  int cx0 = ix0 < 0 ? 0 : ix0;
  int cx1 = (ix0 + 1) > 63 ? 63 : (ix0 + 1);
  int p00 = cy0 * 64 + cx0, p01 = cy0 * 64 + cx1;
  int p10 = cy1 * 64 + cx0, p11 = cy1 * 64 + cx1;
  float w00 = wy0 * wx0, w01 = wy0 * (1.0f - wx0);
  float w10 = (1.0f - wy0) * wx0, w11 = (1.0f - wy0) * (1.0f - wx0);

  int pin = h * 64 + w;
  float r = 0.0f;
  float* op = out + (size_t)idx * T_;
  for (int t = 0; t < T_; ++t) {
    const _Float16* p3 = psp3h + ((size_t)(b * T_ + t)) * OC_ * HW_ + pin;
    float v = 0.0f;
#pragma unroll
    for (int i = 0; i < 8; ++i) v += wu[i] * (float)p3[i * HW_];
    const _Float16* p1 = psp1h + (((size_t)(b * T_ + t)) * CIN_ + o) * HW_;
    v += w00 * (float)p1[p00] + w01 * (float)p1[p01] +
         w10 * (float)p1[p10] + w11 * (float)p1[p11];
    v += r;
    float s = (v >= THETA3) ? 1.0f : 0.0f;
    r = D_REF3 * r - THETA3 * s;
    op[t] = s;
  }
}

// ---------------------------------------------------------------------------
extern "C" void kernel_launch(void* const* d_in, const int* in_sizes, int n_in,
                              void* d_out, int out_size, void* d_ws, size_t ws_size,
                              hipStream_t stream) {
  (void)in_sizes; (void)n_in; (void)out_size; (void)ws_size;
  const float* spikeInput = (const float*)d_in[0];
  const float* w1  = (const float*)d_in[1];
  const float* w2  = (const float*)d_in[2];
  const float* wup = (const float*)d_in[3];
  float* out = (float*)d_out;

  char* ws = (char*)d_ws;
  size_t off = 0;
  auto carve = [&](size_t bytes) -> void* {
    void* p = ws + off;
    off += (bytes + 255) & ~(size_t)255;
    return p;
  };
  _Float16* b1f   = (_Float16*)carve(2 * 32 * 16 * sizeof(_Float16));
  _Float16* b2f   = (_Float16*)carve(3 * 32 * 16 * sizeof(_Float16));
  _Float16* psp1h = (_Float16*)carve((size_t)B_ * T_ * CIN_ * HW_ * sizeof(_Float16));
  float*    ubuf  = (float*)   carve((size_t)B_ * T_ * OC_  * HW_ * sizeof(float));    // u1 == u2
  _Float16* pbuf  = (_Float16*)carve((size_t)B_ * T_ * OC_  * HW_ * sizeof(_Float16)); // psp2 == psp3

  pack_weights<<<1, 192, 0, stream>>>(w1, w2, b1f, b2f);
  psp1_scan<<<(B_ * CIN_ * HW_ + 255) / 256, 256, 0, stream>>>(spikeInput, psp1h);

  int conv_blocks = B_ * T_ * 32;          // 1 block = 1 plane x 2 rows
  conv_wmma<2, 5, 2, 2><<<conv_blocks, 256, 0, stream>>>(psp1h, b1f, ubuf);
  spike_psp<<<(B_ * OC_ * HW_ + 255) / 256, 256, 0, stream>>>(ubuf, pbuf, THETA1, D_REF1, D_SR2);
  conv_wmma<8, 3, 1, 3><<<conv_blocks, 256, 0, stream>>>(pbuf, b2f, ubuf);
  spike_psp<<<(B_ * OC_ * HW_ + 255) / 256, 256, 0, stream>>>(ubuf, pbuf, THETA2, D_REF2, D_SR3);
  out_kernel<<<(B_ * 2 * H2_ * W2_ + 255) / 256, 256, 0, stream>>>(pbuf, psp1h, wup, out);
}
